// DoomLiquidNet_21887153340596
// MI455X (gfx1250) — matrix operand order repaired and checked by reading, not compiled, not run on hardware
//
#include <hip/hip_runtime.h>

// ---------------------------------------------------------------------------
// DoomLiquidNet on MI455X (gfx1250): f16 WMMA implicit-GEMM convs + u-GEMM,
// async global->LDS im2col staging, fp32 VALU liquid-RNN scan. Wave32.
// ---------------------------------------------------------------------------

typedef __attribute__((ext_vector_type(16))) _Float16 v16h;
typedef __attribute__((ext_vector_type(8)))  _Float16 v8h;
typedef __attribute__((ext_vector_type(8)))  float    v8f;

union V16 { v16h v; v8h h[2]; };

#define FEAT  12544
#define BT    1024

// --- WMMA fragment loaders (layouts per CDNA5 ISA 7.12.2, 16-bit A/B) ------
// A: 16x32 (MxK), row-major source with leading dim ld (in halfs).
//   lane<16 : M=lane,    K = {0..7, 16..23}
//   lane>=16: M=lane-16, K = {8..15, 24..31}
__device__ inline v16h loadA16x32(const _Float16* p, int ld, int kc, int lane) {
  int m  = lane & 15;
  int b0 = (lane & 16) ? 8 : 0;
  V16 r;
  r.h[0] = *(const v8h*)(p + (size_t)m * ld + kc + b0);
  r.h[1] = *(const v8h*)(p + (size_t)m * ld + kc + 16 + b0);
  return r.v;
}
// B: 32x16 (KxN), source stored transposed as [n][k] (k contiguous, ldk halfs)
//   lane<16 : N=lane,    K = {0..7, 16..23} ; lane>=16: N=lane-16, K = {8..15, 24..31}
__device__ inline v16h loadBn16(const _Float16* bt, int ldk, int kc, int lane) {
  int n  = lane & 15;
  int kb = (lane & 16) ? 8 : 0;
  V16 r;
  r.h[0] = *(const v8h*)(bt + (size_t)n * ldk + kc + kb);
  r.h[1] = *(const v8h*)(bt + (size_t)n * ldk + kc + kb + 16);
  return r.v;
}

// --- CDNA5 async global->LDS copy (ASYNCcnt-tracked, no VGPR round trip) ---
__device__ inline void async_g2l_b32(void* lds, const void* g) {
  unsigned loff = (unsigned)(unsigned long long)(uintptr_t)lds; // LDS offset = low 32 bits
  asm volatile("global_load_async_to_lds_b32 %0, %1, off"
               :: "v"(loff), "v"(g) : "memory");
}
__device__ inline void wait_async0() {
  asm volatile("s_wait_asynccnt 0x0" ::: "memory");
}

// --- prep kernels ----------------------------------------------------------
__global__ void cvt_f32_f16(const float* __restrict__ src,
                            _Float16* __restrict__ dst, int n) {
  int i = blockIdx.x * blockDim.x + threadIdx.x;
  if (i < n) dst[i] = (_Float16)src[i];
}

// conv1 weights (32,3,4,4) -> f16 [32][64], K padded 48->64 with zeros.
__global__ void prep_w1(const float* __restrict__ w1, _Float16* __restrict__ w1h) {
  int i = blockIdx.x * blockDim.x + threadIdx.x;
  if (i < 32 * 64) {
    int oc = i >> 6, k = i & 63;
    w1h[i] = (k < 48) ? (_Float16)w1[oc * 48 + k] : (_Float16)0.f;
  }
}

// W_in (12544,128) -> transposed f16 winT (128,12544): both GEMM operands of
// the u-GEMM then fragment-load directly from global (no LDS, no barriers).
__global__ void transpose_win(const float* __restrict__ Wbb,
                              _Float16* __restrict__ winT) {
  long i = (long)blockIdx.x * 256 + threadIdx.x;
  if (i < (long)FEAT * 128) {
    int col = (int)(i / FEAT), f = (int)(i % FEAT);
    winT[i] = (_Float16)Wbb[(size_t)f * 128 + col];
  }
}

// --- conv1: implicit GEMM  M=32, K=48(->64), N=1024*30*30 ------------------
// block = 64 threads (2 waves); each block = one 16-column N tile.
__global__ __launch_bounds__(64) void conv1_wmma(
    const float* __restrict__ x,          // (1024,3,62,62) fp32
    const _Float16* __restrict__ w1h,     // [32][64] padded
    const float* __restrict__ b1,
    _Float16* __restrict__ c1h) {         // (1024,32,30,30) f16
  __shared__ _Float16 Btile[16][72];      // [n][k], row pad -> bank spread
  const int tid = threadIdx.x, lane = tid & 31, wv = tid >> 5;
  const int nt = blockIdx.x;              // 0..57599

  for (int e = tid; e < 16 * 64; e += 64) {
    int c = e >> 6, k = e & 63;
    _Float16 val = (_Float16)0.f;
    if (k < 48) {
      int gn = nt * 16 + c;
      int bt = gn / 900, s = gn % 900;
      int y = s / 30, xx = s % 30;
      int ic = k >> 4, r = k & 15, ky = r >> 2, kx = r & 3;
      val = (_Float16)x[((bt * 3 + ic) * 62 + (2 * y + ky)) * 62 + (2 * xx + kx)];
    }
    Btile[c][k] = val;
  }
  __syncthreads();

  v8f acc = {};
  const _Float16* Aw = w1h + wv * 16 * 64;
#pragma unroll
  for (int kk = 0; kk < 2; ++kk) {
    int kc = kk * 32;
    v16h a = loadA16x32(Aw, 64, kc, lane);
    v16h b = loadBn16(&Btile[0][0], 72, kc, lane);
    acc = __builtin_amdgcn_wmma_f32_16x16x32_f16(false, a, false, b,
                                                 (short)0, acc, false, false);
  }
  // epilogue: bias + relu + f16 store
  int c = lane & 15, hi = lane >> 4;
  int gn = nt * 16 + c;
  int bt = gn / 900, s = gn % 900;
#pragma unroll
  for (int r = 0; r < 8; ++r) {
    int oc = wv * 16 + hi * 8 + r;
    float v = acc[r] + b1[oc];
    v = v > 0.f ? v : 0.f;
    c1h[(bt * 32 + oc) * 900 + s] = (_Float16)v;
  }
}

// --- conv2: implicit GEMM  M=64, K=512, N=196 per image --------------------
// block = 128 threads (4 waves): one image x one 16-col tile; wave = oc tile.
// im2col B-tile staged with GLOBAL_LOAD_ASYNC_TO_LDS_B32 (4B-aligned
// contiguous kx-pairs), tracked by ASYNCcnt.
__global__ __launch_bounds__(128) void conv2_wmma(
    const _Float16* __restrict__ c1h,     // (1024,32,30,30)
    const _Float16* __restrict__ w2h,     // [64][512]
    const float* __restrict__ b2,
    _Float16* __restrict__ feat) {        // (1024, 12544) f16
  __shared__ _Float16 Btile[16][520];     // [n][k], row pad -> bank spread
  const int tid = threadIdx.x, lane = tid & 31, wv = tid >> 5;
  const int bt = blockIdx.x, nt = blockIdx.y;

  // 4096 segments of 2 halfs: seg -> (c, ic, ky, kx-pair)
  for (int seg = tid; seg < 16 * 256; seg += 128) {
    int c  = seg >> 8;
    int k2 = (seg & 255) * 2;             // even k: ic*16 + ky*4 + kx0
    int s  = nt * 16 + c;
    if (s < 196) {
      int y = s / 14, xx = s % 14;
      int ic = k2 >> 4, r = k2 & 15, ky = r >> 2, kx0 = r & 3;
      const _Float16* g = c1h + ((size_t)(bt * 32 + ic) * 30 + (2 * y + ky)) * 30
                              + (2 * xx + kx0);
      async_g2l_b32(&Btile[c][k2], g);
    } else {
      *(unsigned*)&Btile[c][k2] = 0u;     // OOB tile columns -> zeros
    }
  }
  wait_async0();
  __syncthreads();

  v8f acc = {};
  const _Float16* Aw = w2h + wv * 16 * 512;
#pragma unroll 4
  for (int kk = 0; kk < 16; ++kk) {
    int kc = kk * 32;
    v16h a = loadA16x32(Aw, 512, kc, lane);
    v16h b = loadBn16(&Btile[0][0], 520, kc, lane);
    acc = __builtin_amdgcn_wmma_f32_16x16x32_f16(false, a, false, b,
                                                 (short)0, acc, false, false);
  }
  int c = lane & 15, hi = lane >> 4;
  int s = nt * 16 + c;
  if (s < 196) {
#pragma unroll
    for (int r = 0; r < 8; ++r) {
      int oc = wv * 16 + hi * 8 + r;
      float v = acc[r] + b2[oc];
      v = v > 0.f ? v : 0.f;
      feat[(size_t)bt * FEAT + oc * 196 + s] = (_Float16)v;
    }
  }
}

// --- u^T = winT @ feat^T : M=128 (cols), N=1024 (rows), K=12544 ------------
// Pre-transposed winT makes BOTH operands direct global fragment loads:
// no LDS, no barriers. block = 256 threads (8 waves) = one 16-row bt tile;
// wave w covers output cols w*16..w*16+15.
__global__ __launch_bounds__(256) void ugemm_wmma(
    const _Float16* __restrict__ feat,    // (1024,12544) [bt][f]
    const _Float16* __restrict__ winT,    // (128,12544)  [col][f]
    const float* __restrict__ bbb,
    float* __restrict__ u) {              // (1024,128) fp32
  const int lane = threadIdx.x & 31, wv = threadIdx.x >> 5;
  const int bt0 = blockIdx.x * 16;
  const _Float16* A  = winT + (size_t)(wv * 16) * FEAT;  // A rows = out col
  const _Float16* Bb = feat + (size_t)bt0 * FEAT;        // B "cols" = bt rows

  v8f acc = {};
  for (int fc = 0; fc < FEAT; fc += 64) {
    v16h a0 = loadA16x32(A, FEAT, fc, lane);
    v16h b0 = loadBn16(Bb, FEAT, fc, lane);
    acc = __builtin_amdgcn_wmma_f32_16x16x32_f16(false, a0, false, b0,
                                                 (short)0, acc, false, false);
    v16h a1 = loadA16x32(A, FEAT, fc + 32, lane);
    v16h b1 = loadBn16(Bb, FEAT, fc + 32, lane);
    acc = __builtin_amdgcn_wmma_f32_16x16x32_f16(false, a1, false, b1,
                                                 (short)0, acc, false, false);
  }
  int n = lane & 15, hi = lane >> 4;
  int bt = bt0 + n;
#pragma unroll
  for (int r = 0; r < 8; ++r) {
    int col = wv * 16 + hi * 8 + r;
    u[(size_t)bt * 128 + col] = acc[r] + bbb[col];
  }
}

// --- liquid RNN scan: one block per batch element, 128 threads -------------
__global__ __launch_bounds__(128) void rnn_kernel(
    const float* __restrict__ u,      // (16,64,128)
    const float* __restrict__ W_bb,   // (12672,128); W_h = rows 12544..
    const float* __restrict__ W_ff1, const float* __restrict__ b_ff1,
    const float* __restrict__ W_ff2, const float* __restrict__ b_ff2,
    const float* __restrict__ W_ta,  const float* __restrict__ b_ta,
    const float* __restrict__ W_tb,  const float* __restrict__ b_tb,
    const float* __restrict__ W_out, const float* __restrict__ b_out,
    float* __restrict__ out) {        // (16,8)
  const int b = blockIdx.x, tid = threadIdx.x;
  __shared__ float hid[64];
  __shared__ float z[128];
  __shared__ float p1[128], p2[128], p3[128];
  if (tid < 64) hid[tid] = 0.f;
  __syncthreads();

  const float* W_h = W_bb + (size_t)FEAT * 128;
  const float* ub  = u + (size_t)b * 64 * 128;

  for (int t = 0; t < 64; ++t) {
    { // z[k] = lecun_tanh(u[t,k] + hid @ W_h)
      int k = tid;
      float s = ub[t * 128 + k];
#pragma unroll 8
      for (int j = 0; j < 64; ++j) s += hid[j] * W_h[j * 128 + k];
      z[k] = 1.7159f * tanhf(0.666f * s);
    }
    __syncthreads();
    { // split-K partials for ff1/ff2/(ta+tb)
      int uu = tid & 63, h = tid >> 6, k0 = h * 64;
      float s1 = 0.f, s2 = 0.f, s3 = 0.f;
#pragma unroll 8
      for (int k = k0; k < k0 + 64; ++k) {
        float zv = z[k];
        s1 += zv * W_ff1[k * 64 + uu];
        s2 += zv * W_ff2[k * 64 + uu];
        s3 += zv * (W_ta[k * 64 + uu] + W_tb[k * 64 + uu]);
      }
      p1[tid] = s1; p2[tid] = s2; p3[tid] = s3;
    }
    __syncthreads();
    float hn = 0.f;
    const bool wr = tid < 64;
    if (wr) {
      int uu = tid;
      float s1 = p1[uu] + p1[uu + 64] + b_ff1[uu];
      float s2 = p2[uu] + p2[uu + 64] + b_ff2[uu];
      float s3 = p3[uu] + p3[uu + 64] + b_ta[uu] + b_tb[uu];
      float f1 = tanhf(s1), f2 = tanhf(s2);
      float ti = 1.f / (1.f + expf(-s3));
      hn = f1 * (1.f - ti) + ti * f2;
    }
    __syncthreads();
    if (wr) hid[tid] = hn;
    __syncthreads();
  }
  if (tid < 8) {
    float s = b_out[tid];
    for (int uu = 0; uu < 64; ++uu) s += hid[uu] * W_out[uu * 8 + tid];
    out[b * 8 + tid] = s;
  }
}

// ---------------------------------------------------------------------------
extern "C" void kernel_launch(void* const* d_in, const int* in_sizes, int n_in,
                              void* d_out, int out_size, void* d_ws, size_t ws_size,
                              hipStream_t stream) {
  (void)in_sizes; (void)n_in; (void)out_size; (void)ws_size;
  const float* x     = (const float*)d_in[0];
  const float* w1    = (const float*)d_in[1];
  const float* b1    = (const float*)d_in[2];
  const float* w2    = (const float*)d_in[3];
  const float* b2    = (const float*)d_in[4];
  const float* W_bb  = (const float*)d_in[5];
  const float* b_bb  = (const float*)d_in[6];
  const float* W_ff1 = (const float*)d_in[7];
  const float* b_ff1 = (const float*)d_in[8];
  const float* W_ff2 = (const float*)d_in[9];
  const float* b_ff2 = (const float*)d_in[10];
  const float* W_ta  = (const float*)d_in[11];
  const float* b_ta  = (const float*)d_in[12];
  const float* W_tb  = (const float*)d_in[13];
  const float* b_tb  = (const float*)d_in[14];
  const float* W_out = (const float*)d_in[15];
  const float* b_out = (const float*)d_in[16];
  float* out = (float*)d_out;

  char* ws = (char*)d_ws;
  size_t off = 0;
  auto alloc = [&](size_t bytes) -> void* {
    void* p = ws + off;
    off = (off + bytes + 255) & ~(size_t)255;
    return p;
  };
  _Float16* c1h  = (_Float16*)alloc((size_t)BT * 32 * 900 * 2);   // 59.0 MB
  _Float16* w1h  = (_Float16*)alloc((size_t)32 * 64 * 2);
  _Float16* w2h  = (_Float16*)alloc((size_t)64 * 512 * 2);
  _Float16* winT = (_Float16*)alloc((size_t)FEAT * 128 * 2);      //  3.2 MB
  _Float16* feat = (_Float16*)alloc((size_t)BT * FEAT * 2);       // 25.7 MB
  float*    u    = (float*)alloc((size_t)BT * 128 * 4);           //  0.5 MB

  prep_w1<<<8, 256, 0, stream>>>(w1, w1h);
  cvt_f32_f16<<<(64 * 512 + 255) / 256, 256, 0, stream>>>(w2, w2h, 64 * 512);
  transpose_win<<<(FEAT * 128 + 255) / 256, 256, 0, stream>>>(W_bb, winT);

  conv1_wmma<<<57600, 64, 0, stream>>>(x, w1h, b1, c1h);
  conv2_wmma<<<dim3(BT, 13), 128, 0, stream>>>(c1h, w2h, b2, feat);
  ugemm_wmma<<<64, 256, 0, stream>>>(feat, winT, b_bb, u);
  rnn_kernel<<<16, 128, 0, stream>>>(u, W_bb, W_ff1, b_ff1, W_ff2, b_ff2,
                                     W_ta, b_ta, W_tb, b_tb, W_out, b_out, out);
}
